// PointTransformerCls_11020886082150
// MI455X (gfx1250) — matrix operand order repaired and checked by reading, not compiled, not run on hardware
//
#include <hip/hip_runtime.h>
#include <hip/hip_bf16.h>

#define BN_EPS 1e-5f
#define BATCH 16

typedef __attribute__((ext_vector_type(16))) _Float16 v16h;
typedef __attribute__((ext_vector_type(8)))  _Float16 v8h;
typedef __attribute__((ext_vector_type(8)))  float    v8f;

// ---------------------------------------------------------------------------
// Generic small kernels
// ---------------------------------------------------------------------------
__global__ void k_linear_rows(const float* __restrict__ in, const float* __restrict__ W,
                              const float* __restrict__ bias, float* __restrict__ out,
                              int rows, int Cin, int Cout) {
  int i = blockIdx.x * blockDim.x + threadIdx.x;
  if (i >= rows * Cout) return;
  int r = i / Cout, d = i % Cout;
  float acc = bias ? bias[d] : 0.f;
  const float* ir = in + (size_t)r * Cin;
  for (int c = 0; c < Cin; ++c) acc += ir[c] * W[c * Cout + d];
  out[i] = acc;
}

// per-channel sum / sumsq over rows  (one block per channel)
__global__ void k_channel_stats(const float* __restrict__ src, int rows, int C,
                                float* __restrict__ sum, float* __restrict__ sumsq) {
  __shared__ float ss[256], sq[256];
  int d = blockIdx.x, t = threadIdx.x;
  float a = 0.f, b = 0.f;
  for (int r = t; r < rows; r += blockDim.x) {
    float v = src[(size_t)r * C + d];
    a += v; b += v * v;
  }
  ss[t] = a; sq[t] = b; __syncthreads();
  for (int s = blockDim.x / 2; s > 0; s >>= 1) {
    if (t < s) { ss[t] += ss[t + s]; sq[t] += sq[t + s]; }
    __syncthreads();
  }
  if (t == 0) { sum[d] = ss[0]; sumsq[d] = sq[0]; }
}

__global__ void k_bn_finalize(const float* __restrict__ sum, const float* __restrict__ sumsq,
                              const float* __restrict__ g, const float* __restrict__ bta,
                              float invcount, int C,
                              float* __restrict__ scale, float* __restrict__ shift) {
  int d = blockIdx.x * blockDim.x + threadIdx.x;
  if (d >= C) return;
  float mean = sum[d] * invcount;
  float var  = sumsq[d] * invcount - mean * mean;
  float sc   = g[d] * rsqrtf(var + BN_EPS);
  scale[d] = sc;
  shift[d] = bta[d] - mean * sc;
}

__global__ void k_bn_apply_relu(float* __restrict__ buf, const float* __restrict__ scale,
                                const float* __restrict__ shift, int n, int C) {
  int i = blockIdx.x * blockDim.x + threadIdx.x;
  if (i >= n) return;
  int d = i % C;
  float v = buf[i] * scale[d] + shift[d];
  buf[i] = v > 0.f ? v : 0.f;
}

// ---------------------------------------------------------------------------
// Farthest point sampling: one block per batch, mind[] lives in LDS
// ---------------------------------------------------------------------------
__global__ void k_fps(const float* __restrict__ p, int* __restrict__ idx, int N, int M) {
  __shared__ float mind[4096];
  __shared__ float bv[256];
  __shared__ int   bi[256];
  __shared__ float lp[3];
  __shared__ int   s_last;
  int b = blockIdx.x, t = threadIdx.x;
  const float* pb = p + (size_t)b * N * 3;
  for (int j = t; j < N; j += blockDim.x) mind[j] = 3.4e38f;
  if (t == 0) s_last = 0;
  __syncthreads();
  for (int i = 0; i < M; ++i) {
    int last = s_last;
    if (t == 0) {
      idx[b * M + i] = last;
      lp[0] = pb[last * 3 + 0]; lp[1] = pb[last * 3 + 1]; lp[2] = pb[last * 3 + 2];
    }
    __syncthreads();
    float lx = lp[0], ly = lp[1], lz = lp[2];
    float best = -1.f; int besti = 0;
    for (int j = t; j < N; j += blockDim.x) {
      float dx = pb[j * 3] - lx, dy = pb[j * 3 + 1] - ly, dz = pb[j * 3 + 2] - lz;
      float d = dx * dx + dy * dy + dz * dz;
      float mv = mind[j]; mv = d < mv ? d : mv; mind[j] = mv;
      if (mv > best) { best = mv; besti = j; }
    }
    bv[t] = best; bi[t] = besti; __syncthreads();
    for (int s = blockDim.x / 2; s > 0; s >>= 1) {
      if (t < s) {
        if (bv[t + s] > bv[t] || (bv[t + s] == bv[t] && bi[t + s] < bi[t])) {
          bv[t] = bv[t + s]; bi[t] = bi[t + s];
        }
      }
      __syncthreads();
    }
    if (t == 0) s_last = bi[0];
    __syncthreads();
  }
}

__global__ void k_gather_points(const float* __restrict__ p, const int* __restrict__ idx,
                                float* __restrict__ pnew, int N, int M) {
  int i = blockIdx.x * blockDim.x + threadIdx.x;
  if (i >= BATCH * M * 3) return;
  int c = i % 3, bm = i / 3, b = bm / M;
  pnew[i] = p[((size_t)b * N + idx[bm]) * 3 + c];
}

// kNN-16 (smallest squared distance, stable tie-break) — one thread per center
__global__ void k_knn(const float* __restrict__ p, const float* __restrict__ pnew,
                      int* __restrict__ nidx, int N, int M) {
  int i = blockIdx.x * blockDim.x + threadIdx.x;
  if (i >= BATCH * M) return;
  int b = i / M;
  const float* pb = p + (size_t)b * N * 3;
  float qx = pnew[i * 3], qy = pnew[i * 3 + 1], qz = pnew[i * 3 + 2];
  float bd[16]; int bix[16];
#pragma unroll
  for (int k = 0; k < 16; ++k) { bd[k] = 3.4e38f; bix[k] = 0; }
  for (int j = 0; j < N; ++j) {
    float dx = pb[j * 3] - qx, dy = pb[j * 3 + 1] - qy, dz = pb[j * 3 + 2] - qz;
    float d = dx * dx + dy * dy + dz * dz;
    if (d < bd[15]) {
      int pos = 15;
#pragma unroll
      for (int k = 14; k >= 0; --k) if (d < bd[k]) pos = k;
#pragma unroll
      for (int k = 15; k > 0; --k) if (k > pos) { bd[k] = bd[k - 1]; bix[k] = bix[k - 1]; }
      bd[pos] = d; bix[pos] = j;
    }
  }
#pragma unroll
  for (int k = 0; k < 16; ++k) nidx[i * 16 + k] = bix[k];
}

__global__ void k_meanpool(const float* __restrict__ h, float* __restrict__ z, int M, int C) {
  int i = blockIdx.x * blockDim.x + threadIdx.x;
  if (i >= BATCH * C) return;
  int b = i / C, d = i % C;
  float acc = 0.f;
  for (int m = 0; m < M; ++m) acc += h[((size_t)b * M + m) * C + d];
  z[i] = acc / (float)M;
}

// W[c][d] (f32) -> Wt[d][c] (f16, c padded to CINP with zeros)
__global__ void k_w_to_f16t(const float* __restrict__ W, _Float16* __restrict__ Wt,
                            int CIN, int CINP, int D) {
  int i = blockIdx.x * blockDim.x + threadIdx.x;
  if (i >= D * CINP) return;
  int d = i / CINP, c = i % CINP;
  Wt[i] = (c < CIN) ? (_Float16)W[c * D + d] : (_Float16)0.f;
}

// ---------------------------------------------------------------------------
// Grouped gather-GEMM on WMMA.
// One block = one (b,m) group (16 neighbor rows). 128 threads (4 waves).
// Phase 1: cooperatively stage the f16 feature tile At[16][CINP] in LDS
//          (feat = [p[nid]-pnew || h[nid]], zero-padded).
// Phase 2: each wave owns column tiles ct = wv, wv+4, ...; per K-chunk of 32:
//          A fragment = 2x ds_load_b128 from At, B fragment = 2x
//          global_load_b128 from pre-transposed f16 weights, then
//          v_wmma_f32_16x16x32_f16 (f32 accumulate).
// pass 0: per-channel sum/sumsq (BN stats);  pass 1: relu(scale*y+shift),
//         max over the 16 neighbors folded with bitwise atomicMax (vals >= 0).
// ---------------------------------------------------------------------------
template <int CIN, int CINH, int D>
__global__ void k_group_gemm_wmma(int pass,
                                  const float* __restrict__ p_in,
                                  const float* __restrict__ h_in,
                                  const float* __restrict__ pnew,
                                  const int* __restrict__ nidx,
                                  const _Float16* __restrict__ Wt,   // [D][CINP]
                                  int N, int M,
                                  float* __restrict__ sum, float* __restrict__ sumsq,
                                  const float* __restrict__ scale,
                                  const float* __restrict__ shift,
                                  float* __restrict__ h_out) {
  constexpr int CINP   = (CIN + 31) & ~31;
  constexpr int CHUNKS = CINP / 32;
  __shared__ __align__(16) _Float16 At[16][CINP];

  const int bm = blockIdx.x;              // (b,m) group index; rows bm*16..bm*16+15
  const int b  = bm / M;
  const int t  = threadIdx.x;

  // ---- phase 1: stage feature tile in LDS --------------------------------
  {
    int tr = t & 15;                      // neighbor row 0..15
    int tc = t >> 4;                      // column lane 0..7
    int nid = nidx[bm * 16 + tr];
    const float* psrc = p_in + ((size_t)b * N + nid) * 3;
    const float* pq   = pnew + (size_t)bm * 3;
    const float* hsrc = h_in + ((size_t)b * N + nid) * CINH;
    for (int c = tc; c < CINP; c += 8) {
      float v = 0.f;
      if (c < 3)          v = psrc[c] - pq[c];
      else if (c < CIN)   v = hsrc[c - 3];
      At[tr][c] = (_Float16)v;
    }
  }
  __syncthreads();

  // ---- phase 2: WMMA over column tiles -----------------------------------
  const int wv   = t >> 5;                // wave in block (0..3)
  const int lane = t & 31;
  const int hf   = lane >> 4;
  const int lr   = lane & 15;
  const int colTiles = D / 16;

  for (int ct = wv; ct < colTiles; ct += 4) {
    int col = ct * 16 + lr;
    const _Float16* bcol = Wt + (size_t)col * CINP;
    // speculative prefetch of the next column tile's B data (dropped if OOB)
    __builtin_prefetch(bcol + (size_t)64 * CINP, 0, 0);

    v8f acc = {0.f, 0.f, 0.f, 0.f, 0.f, 0.f, 0.f, 0.f};
#pragma unroll
    for (int ch = 0; ch < CHUNKS; ++ch) {
      // A fragment (ISA 16-bit A 16x32 layout): K = 8*hf + 0..7, 16+8*hf + 0..7
      const _Float16* arow = &At[lr][ch * 32 + 8 * hf];
      v8h alo = *(const v8h*)(arow);
      v8h ahi = *(const v8h*)(arow + 16);
      v16h a  = __builtin_shufflevector(alo, ahi, 0, 1, 2, 3, 4, 5, 6, 7,
                                        8, 9, 10, 11, 12, 13, 14, 15);
      // B fragment: K = 8*hf + 0..7, 16+8*hf + 0..7 for column `col` (matches A)
      const _Float16* brow = bcol + ch * 32 + 8 * hf;
      v8h blo = *(const v8h*)(brow);
      v8h bhi = *(const v8h*)(brow + 16);
      v16h bb = __builtin_shufflevector(blo, bhi, 0, 1, 2, 3, 4, 5, 6, 7,
                                        8, 9, 10, 11, 12, 13, 14, 15);
      acc = __builtin_amdgcn_wmma_f32_16x16x32_f16(false, a, false, bb,
                                                   (short)0, acc, false, false);
    }

    if (pass == 0) {
      float s = 0.f, q = 0.f;
#pragma unroll
      for (int v = 0; v < 8; ++v) { float x = acc[v]; s += x; q += x * x; }
      atomicAdd(&sum[col], s);            // all 8 outputs share channel `col`
      atomicAdd(&sumsq[col], q);
    } else {
      float sc = scale[col], sh = shift[col];
      float best = 0.f;
#pragma unroll
      for (int v = 0; v < 8; ++v) {       // all 8 rows belong to this (b,m) group
        float x = acc[v] * sc + sh;
        x = x > 0.f ? x : 0.f;
        best = x > best ? x : best;
      }
      // relu -> non-negative, IEEE bits monotone: bitwise max is float max
      atomicMax((int*)&h_out[(size_t)bm * D + col], __float_as_int(best));
    }
  }
}

// ---------------------------------------------------------------------------
// One transition-down stage
// ---------------------------------------------------------------------------
template <int CIN, int CINH, int D>
static void run_stage(const float* p_in, const float* h_in, float* p_out, float* h_out,
                      const float* W, const float* g, const float* bta, int N, int M,
                      int* fpsidx, int* nidx, _Float16* Wt,
                      float* sum, float* sumsq, float* scale, float* shift,
                      hipStream_t stream) {
  constexpr int CINP = (CIN + 31) & ~31;
  k_fps<<<BATCH, 256, 0, stream>>>(p_in, fpsidx, N, M);
  int nm = BATCH * M;
  k_gather_points<<<(nm * 3 + 255) / 256, 256, 0, stream>>>(p_in, fpsidx, p_out, N, M);
  k_knn<<<(nm + 127) / 128, 128, 0, stream>>>(p_in, p_out, nidx, N, M);
  k_w_to_f16t<<<(D * CINP + 255) / 256, 256, 0, stream>>>(W, Wt, CIN, CINP, D);
  hipMemsetAsync(sum, 0, D * sizeof(float), stream);
  hipMemsetAsync(sumsq, 0, D * sizeof(float), stream);
  int R = nm * 16;
  k_group_gemm_wmma<CIN, CINH, D><<<nm, 128, 0, stream>>>(
      0, p_in, h_in, p_out, nidx, Wt, N, M, sum, sumsq, scale, shift, h_out);
  k_bn_finalize<<<(D + 63) / 64, 64, 0, stream>>>(sum, sumsq, g, bta,
                                                  1.f / (float)R, D, scale, shift);
  hipMemsetAsync(h_out, 0, (size_t)nm * D * sizeof(float), stream);
  k_group_gemm_wmma<CIN, CINH, D><<<nm, 128, 0, stream>>>(
      1, p_in, h_in, p_out, nidx, Wt, N, M, sum, sumsq, scale, shift, h_out);
}

// ---------------------------------------------------------------------------
extern "C" void kernel_launch(void* const* d_in, const int* in_sizes, int n_in,
                              void* d_out, int out_size, void* d_ws, size_t ws_size,
                              hipStream_t stream) {
  (void)in_sizes; (void)n_in; (void)out_size; (void)ws_size;
  const float* x   = (const float*)d_in[0];
  const float* W1  = (const float*)d_in[1];
  const float* g1  = (const float*)d_in[2];
  const float* b1  = (const float*)d_in[3];
  const float* W2  = (const float*)d_in[4];
  const float* g2  = (const float*)d_in[5];
  const float* b2  = (const float*)d_in[6];
  const float* W3  = (const float*)d_in[7];
  const float* g3  = (const float*)d_in[8];
  const float* b3  = (const float*)d_in[9];
  const float* W4  = (const float*)d_in[10];
  const float* g4  = (const float*)d_in[11];
  const float* b4  = (const float*)d_in[12];
  const float* W5  = (const float*)d_in[13];
  const float* g5  = (const float*)d_in[14];
  const float* b5  = (const float*)d_in[15];
  const float* Wc1 = (const float*)d_in[16];
  const float* bc1 = (const float*)d_in[17];
  const float* gc1 = (const float*)d_in[18];
  const float* hc1 = (const float*)d_in[19];
  const float* Wc2 = (const float*)d_in[20];
  const float* bc2 = (const float*)d_in[21];
  const float* gc2 = (const float*)d_in[22];
  const float* hc2 = (const float*)d_in[23];
  const float* Wc3 = (const float*)d_in[24];
  const float* bc3 = (const float*)d_in[25];

  // workspace carve-up (floats)
  float* ws = (float*)d_ws;
  size_t o = 0;
  float* h0 = ws + o; o += (size_t)BATCH * 4096 * 32;
  float* h1 = ws + o; o += (size_t)BATCH * 1024 * 64;
  float* h2 = ws + o; o += (size_t)BATCH * 256 * 128;
  float* h3 = ws + o; o += (size_t)BATCH * 64 * 256;
  float* h4 = ws + o; o += (size_t)BATCH * 16 * 512;
  float* p1 = ws + o; o += (size_t)BATCH * 1024 * 3 + 64;
  float* p2 = ws + o; o += (size_t)BATCH * 256 * 3 + 64;
  float* p3 = ws + o; o += (size_t)BATCH * 64 * 3 + 64;
  float* p4 = ws + o; o += (size_t)BATCH * 16 * 3 + 64;
  int*   fpsidx = (int*)(ws + o); o += (size_t)BATCH * 1024;
  int*   nidx   = (int*)(ws + o); o += (size_t)BATCH * 1024 * 16;
  _Float16* Wt  = (_Float16*)(ws + o); o += (size_t)512 * 288 / 2 + 64;  // max D*CINP halves
  float* sum   = ws + o; o += 512;
  float* sumsq = ws + o; o += 512;
  float* scale = ws + o; o += 512;
  float* shift = ws + o; o += 512;
  float* z  = ws + o; o += (size_t)BATCH * 512;
  float* t1 = ws + o; o += (size_t)BATCH * 256;
  float* t2 = ws + o; o += (size_t)BATCH * 128;

  // --- stem: h0 = relu(bn(x @ W1)) over axes (B,N) ---------------------------
  const int R0 = BATCH * 4096;
  k_linear_rows<<<(R0 * 32 + 255) / 256, 256, 0, stream>>>(x, W1, nullptr, h0, R0, 3, 32);
  k_channel_stats<<<32, 256, 0, stream>>>(h0, R0, 32, sum, sumsq);
  k_bn_finalize<<<1, 64, 0, stream>>>(sum, sumsq, g1, b1, 1.f / (float)R0, 32, scale, shift);
  k_bn_apply_relu<<<(R0 * 32 + 255) / 256, 256, 0, stream>>>(h0, scale, shift, R0 * 32, 32);

  // --- transition-down stages (WMMA grouped GEMM) ----------------------------
  run_stage<35, 32, 64>(x, h0, p1, h1, W2, g2, b2, 4096, 1024,
                        fpsidx, nidx, Wt, sum, sumsq, scale, shift, stream);
  run_stage<67, 64, 128>(p1, h1, p2, h2, W3, g3, b3, 1024, 256,
                         fpsidx, nidx, Wt, sum, sumsq, scale, shift, stream);
  run_stage<131, 128, 256>(p2, h2, p3, h3, W4, g4, b4, 256, 64,
                           fpsidx, nidx, Wt, sum, sumsq, scale, shift, stream);
  run_stage<259, 256, 512>(p3, h3, p4, h4, W5, g5, b5, 64, 16,
                           fpsidx, nidx, Wt, sum, sumsq, scale, shift, stream);

  // --- head ------------------------------------------------------------------
  k_meanpool<<<(BATCH * 512 + 255) / 256, 256, 0, stream>>>(h4, z, 16, 512);

  k_linear_rows<<<(BATCH * 256 + 255) / 256, 256, 0, stream>>>(z, Wc1, bc1, t1, BATCH, 512, 256);
  k_channel_stats<<<256, 256, 0, stream>>>(t1, BATCH, 256, sum, sumsq);
  k_bn_finalize<<<4, 64, 0, stream>>>(sum, sumsq, gc1, hc1, 1.f / (float)BATCH, 256, scale, shift);
  k_bn_apply_relu<<<(BATCH * 256 + 255) / 256, 256, 0, stream>>>(t1, scale, shift, BATCH * 256, 256);

  k_linear_rows<<<(BATCH * 128 + 255) / 256, 256, 0, stream>>>(t1, Wc2, bc2, t2, BATCH, 256, 128);
  k_channel_stats<<<128, 256, 0, stream>>>(t2, BATCH, 128, sum, sumsq);
  k_bn_finalize<<<2, 64, 0, stream>>>(sum, sumsq, gc2, hc2, 1.f / (float)BATCH, 128, scale, shift);
  k_bn_apply_relu<<<(BATCH * 128 + 255) / 256, 256, 0, stream>>>(t2, scale, shift, BATCH * 128, 128);

  k_linear_rows<<<(BATCH * 40 + 255) / 256, 256, 0, stream>>>(t2, Wc3, bc3, (float*)d_out, BATCH, 128, 40);
}
